// ScaledDotProductAttention_42451456754047
// MI455X (gfx1250) — compile-verified, matrix operand order checked by reading, not run on hardware
//
#include <hip/hip_runtime.h>

// CDNA5 / gfx1250 flash-attention (fused scaled-dot-product attention).
// B=16, Sq=Sk=2048, d=64. fp32 in/out, f16 WMMA compute, fp32 online softmax
// in base-2 domain; denominator via ones-column WMMA; row reductions via DPP16;
// all LDS fragment loads are 16B-aligned ds_load_b128.

typedef __attribute__((ext_vector_type(16))) _Float16 v16h;
typedef __attribute__((ext_vector_type(2)))  __fp16   v2fp16;
typedef __attribute__((ext_vector_type(8)))  float    v8f;

#define B_    16
#define SQ    2048
#define SK    2048
#define D_    64
#define QTILE 16      // query rows per wave
#define WAVES 4       // waves per block
#define KT    64      // keys per tile iteration
#define KSTR  72      // LDS stride (halves); 144B => every frag group 16B-aligned
#define VSTR  72
#define PSTR  72

union Frag { v16h h; unsigned int u[8]; uint4 q[2]; };
union HPair { v2fp16 h; unsigned int u; };

// Pack two f32 -> packed f16 dword (v_cvt_pk_rtz_f16_f32).
__device__ __forceinline__ unsigned int pkh2(float x, float y) {
    HPair p; p.h = __builtin_amdgcn_cvt_pkrtz(x, y);
    return p.u;
}

// Raw v_exp_f32 (args here are always <= 0 and far from denormal trouble;
// -1e30 -> 0 is exactly the behavior we want for masked columns).
__device__ __forceinline__ float fexp2(float x) {
#if __has_builtin(__builtin_amdgcn_exp2f)
    return __builtin_amdgcn_exp2f(x);
#else
    return exp2f(x);
#endif
}

// Cross-lane move via DPP16 (single VALU slot, no LDS traffic).
template <int CTRL>
__device__ __forceinline__ float dppf(float x) {
    return __int_as_float(__builtin_amdgcn_update_dpp(
        0, __float_as_int(x), CTRL, 0xf, 0xf, true));
}

// All-reduce max over each 16-lane DPP row (our two lane halves are disjoint
// row halves of the D fragment, so this is exactly the row-wise max).
__device__ __forceinline__ float rowmax16(float v) {
    v = fmaxf(v, dppf<0xB1>(v));    // quad_perm [1,0,3,2]  (xor 1)
    v = fmaxf(v, dppf<0x4E>(v));    // quad_perm [2,3,0,1]  (xor 2)
    v = fmaxf(v, dppf<0x141>(v));   // row_half_mirror      (xor 7 -> quads)
    v = fmaxf(v, dppf<0x140>(v));   // row_mirror           (xor 15 -> octets)
    return v;
}

// Load one 16x32 f16 A/B fragment from a 16B-aligned LDS row base
// (base already includes the row offset and this lane's hi*8 K-group shift).
__device__ __forceinline__ void ldfrag(Frag& fr, const _Float16* base) {
    fr.q[0] = *(const uint4*)(base);        // K pairs 0..7   (this half)
    fr.q[1] = *(const uint4*)(base + 16);   // K pairs 16..23 (this half)
}

__global__ __launch_bounds__(128)
void fa_fwd_kernel(const float* __restrict__ Q,
                   const float* __restrict__ K,
                   const float* __restrict__ V,
                   const int*   __restrict__ mask,   // [B, Sk], nonzero = padded
                   float* __restrict__ O)
{
    __shared__ unsigned int sK32[KT * KSTR / 2];          // K tile [key][d], f16
    __shared__ unsigned int sV32[D_ * VSTR / 2];          // V^T tile [d][key], f16
    __shared__ unsigned int sP32[WAVES * QTILE * PSTR / 2];

    _Float16* const sKh = (_Float16*)sK32;
    _Float16* const sVh = (_Float16*)sV32;
    _Float16* const sPh = (_Float16*)sP32;

    const int tid  = threadIdx.x;
    const int wave = tid >> 5;
    const int lane = tid & 31;
    const int n    = lane & 15;          // fragment column / row-within-16
    const int hi   = (lane >> 4) & 1;    // lane half selects K/M sub-range

    const int b     = blockIdx.x >> 5;           // SQ/(QTILE*WAVES)=32 chunks
    const int qc    = blockIdx.x & 31;
    const int qbase = qc * (QTILE * WAVES) + wave * QTILE;
    const int bSK   = b * SK;

    // Loop-invariant per-lane staging offsets (float units).
    const int kRow0 = tid >> 4;                  // K stage: base row, +8 per iter
    const int kOff0 = kRow0 * D_ + ((tid & 15) << 2);
    const int vRow0 = (tid >> 5) << 1;           // V stage: base row pair, +8 per iter
    const int vCol0 = (tid & 31) << 1;
    const int vOff0 = vRow0 * D_ + vCol0;

    // ---- Q A-fragments (16x64 -> two chained 16x32).
    // Scale folds 1/sqrt(d) and log2(e) so softmax runs in base-2 domain.
    const float qscale = 0.125f * 1.44269504088896340736f;
    Frag qa0, qa1;
    {
        const float* qrow = Q + (size_t)(b * SQ + qbase + n) * D_;
        #pragma unroll
        for (int v = 0; v < 8; ++v) {
            const int kb = ((v >= 4) ? 16 : 0) + (hi ? 8 : 0) + ((v & 3) << 1);
            qa0.u[v] = pkh2(qrow[kb]      * qscale, qrow[kb + 1]      * qscale);
            qa1.u[v] = pkh2(qrow[32 + kb] * qscale, qrow[32 + kb + 1] * qscale);
        }
    }

    // Constant all-ones B fragment: accumulates softmax denominator via WMMA.
    Frag ones;
    #pragma unroll
    for (int v = 0; v < 8; ++v) ones.u[v] = 0x3C003C00u;   // (1.0h, 1.0h)

    // Loop-invariant LDS fragment base pointers (include hi*8 K-group shift).
    _Float16* const sPw  = sPh + wave * (QTILE * PSTR);
    const _Float16* kfrow[4];
    const _Float16* vcrow[4];
    #pragma unroll
    for (int f = 0; f < 4; ++f) kfrow[f] = sKh + (16 * f + n) * KSTR + 8 * hi;
    #pragma unroll
    for (int c = 0; c < 4; ++c) vcrow[c] = sVh + (c * 16 + n) * VSTR + 8 * hi;
    const _Float16* const prow = sPw + n * PSTR + 8 * hi;
    unsigned short* const pst16 = (unsigned short*)(sPw + (8 * hi) * PSTR + n);

    // Online-softmax state: this lane touches rows (r + 8*hi), r=0..7.
    // oacc[0..3] = output chunks, oacc[4] = running row-sum (denominator).
    float m_run[8];
    v8f   oacc[5] = {v8f{}, v8f{}, v8f{}, v8f{}, v8f{}};
    #pragma unroll
    for (int r = 0; r < 8; ++r) m_run[r] = -1e30f;

    #pragma clang loop unroll(disable)
    for (int kt = 0; kt < SK / KT; ++kt) {
        const int key0 = kt * KT;
        const float* const Kt = K + (size_t)(bSK + key0) * D_;   // uniform bases
        const float* const Vt = V + (size_t)(bSK + key0) * D_;

        // ---- cooperative stage: K tile row-major f16 (packed dword stores)
        #pragma unroll
        for (int i = 0; i < 8; ++i) {
            const float4 kv = *(const float4*)(Kt + kOff0 + i * (8 * D_));
            const int idx = ((kRow0 + 8 * i) * KSTR + ((tid & 15) << 2)) >> 1;
            sK32[idx]     = pkh2(kv.x, kv.y);
            sK32[idx + 1] = pkh2(kv.z, kv.w);
        }
        // ---- V staged transposed in 2x2 blocks so LDS writes are packed dwords
        #pragma unroll
        for (int i = 0; i < 8; ++i) {
            const float2 va = *(const float2*)(Vt + vOff0 + i * (8 * D_));
            const float2 vb = *(const float2*)(Vt + vOff0 + i * (8 * D_) + D_);
            const int r2 = vRow0 + 8 * i;
            sV32[(vCol0 * VSTR + r2) >> 1]       = pkh2(va.x, vb.x);
            sV32[((vCol0 + 1) * VSTR + r2) >> 1] = pkh2(va.y, vb.y);
        }
        __syncthreads();

        // ---- scores: four 16x16 fragments over 64 keys
        v8f s[4];
        #pragma unroll
        for (int f = 0; f < 4; ++f) {
            Frag b0, b1;
            ldfrag(b0, kfrow[f]);
            ldfrag(b1, kfrow[f] + 32);
            v8f acc = {};
            acc = __builtin_amdgcn_wmma_f32_16x16x32_f16(false, qa0.h, false, b0.h, (short)0, acc, false, false);
            acc = __builtin_amdgcn_wmma_f32_16x16x32_f16(false, qa1.h, false, b1.h, (short)0, acc, false, false);
            s[f] = acc;
        }

        // ---- key padding mask: column index == n for both lane halves
        const int* mrow = mask + bSK + key0;
        #pragma unroll
        for (int f = 0; f < 4; ++f) {
            if (mrow[16 * f + n] != 0) {
                #pragma unroll
                for (int r = 0; r < 8; ++r) s[f][r] = -1e30f;
            }
        }

        // ---- online softmax update (row max via DPP16, no LDS traffic)
        #pragma unroll
        for (int r = 0; r < 8; ++r) {
            const float v = rowmax16(
                fmaxf(fmaxf(s[0][r], s[1][r]), fmaxf(s[2][r], s[3][r])));
            const float mnew  = fmaxf(m_run[r], v);
            const float alpha = fexp2(m_run[r] - mnew);
            m_run[r] = mnew;
            s[0][r] = fexp2(s[0][r] - mnew);
            s[1][r] = fexp2(s[1][r] - mnew);
            s[2][r] = fexp2(s[2][r] - mnew);
            s[3][r] = fexp2(s[3][r] - mnew);
            oacc[0][r] *= alpha;
            oacc[1][r] *= alpha;
            oacc[2][r] *= alpha;
            oacc[3][r] *= alpha;
            oacc[4][r] *= alpha;    // denominator rescales identically
        }

        // ---- re-swizzle P: D-layout (f32) -> LDS -> A-layout (f16), wave-local
        // Packed convert; >>16 truncating store selects ds_store_b16_d16_hi.
        #pragma unroll
        for (int r = 0; r < 8; ++r) {
            const unsigned int u01 = pkh2(s[0][r], s[1][r]);
            const unsigned int u23 = pkh2(s[2][r], s[3][r]);
            pst16[r * PSTR +  0] = (unsigned short)u01;
            pst16[r * PSTR + 16] = (unsigned short)(u01 >> 16);
            pst16[r * PSTR + 32] = (unsigned short)u23;
            pst16[r * PSTR + 48] = (unsigned short)(u23 >> 16);
        }
        Frag pa0, pa1;
        ldfrag(pa0, prow);
        ldfrag(pa1, prow + 32);

        // ---- O += P (16x64) x V (64x16) per 16-wide d chunk, K=64 chained
        #pragma unroll
        for (int c = 0; c < 4; ++c) {
            Frag vb0, vb1;
            ldfrag(vb0, vcrow[c]);
            ldfrag(vb1, vcrow[c] + 32);
            oacc[c] = __builtin_amdgcn_wmma_f32_16x16x32_f16(false, pa0.h, false, vb0.h, (short)0, oacc[c], false, false);
            oacc[c] = __builtin_amdgcn_wmma_f32_16x16x32_f16(false, pa1.h, false, vb1.h, (short)0, oacc[c], false, false);
        }
        // ---- softmax denominator via WMMA against ones-column
        oacc[4] = __builtin_amdgcn_wmma_f32_16x16x32_f16(false, pa0.h, false, ones.h, (short)0, oacc[4], false, false);
        oacc[4] = __builtin_amdgcn_wmma_f32_16x16x32_f16(false, pa1.h, false, ones.h, (short)0, oacc[4], false, false);

        __syncthreads();   // protect sK/sV before next tile's stage
    }

    // ---- normalize and write out (D layout: row = r + 8*hi, col = c*16 + n)
    #pragma unroll
    for (int r = 0; r < 8; ++r) {
        const float inv = 1.0f / oacc[4][r];
        float* orow = O + (size_t)(b * SQ + qbase + r + 8 * hi) * D_;
        orow[ 0 + n] = oacc[0][r] * inv;
        orow[16 + n] = oacc[1][r] * inv;
        orow[32 + n] = oacc[2][r] * inv;
        orow[48 + n] = oacc[3][r] * inv;
    }
}

extern "C" void kernel_launch(void* const* d_in, const int* in_sizes, int n_in,
                              void* d_out, int out_size, void* d_ws, size_t ws_size,
                              hipStream_t stream) {
    const float* Q    = (const float*)d_in[0];
    const float* K    = (const float*)d_in[1];
    const float* V    = (const float*)d_in[2];
    const int*   mask = (const int*)d_in[3];
    float*       O    = (float*)d_out;
    (void)in_sizes; (void)n_in; (void)out_size; (void)d_ws; (void)ws_size;

    const int grid = B_ * (SQ / (QTILE * WAVES));   // 16 * 32 = 512 blocks
    fa_fwd_kernel<<<grid, WAVES * 32, 0, stream>>>(Q, K, V, mask, O);
}
